// MultiHeadAttention_30442728194370
// MI455X (gfx1250) — matrix operand order repaired and checked by reading, not compiled, not run on hardware
//
#include <hip/hip_runtime.h>

// ---------------------------------------------------------------------------
// Fused MHA-over-heads for MI455X (gfx1250, wave32, WMMA bf16 16x16x32).
//   d_in: q,k,v [4,4096,1024] f32; Wq,bq,Wk,bk,Wv,bv,Wo,bo f32
//   d_ws: 8 MB -> four transposed bf16 weights Wt[n][k]
//   One workgroup = 32 token rows (512 thr / 16 waves); Q/K/V/attn never
//   leave LDS (272 KB). M=32 halves L2 weight traffic vs M=16 and lets each
//   B fragment feed two WMMAs.
// ---------------------------------------------------------------------------

typedef __bf16 bf16_t;
typedef __attribute__((ext_vector_type(16))) __bf16 v16bf;
typedef __attribute__((ext_vector_type(8)))  __bf16 v8bf;
typedef __attribute__((ext_vector_type(8)))  float  v8f;

#define D_MODEL 1024
#define HDIM    64
#define ROWS_PER_WG 32
#define THREADS 512           // 16 waves; wave w owns 64 output columns

__device__ __forceinline__ v8f wmma_bf16(v16bf a, v16bf b, v8f c) {
  // D = A(16x32 bf16) * B(32x16 bf16) + C(16x16 f32)
  return __builtin_amdgcn_wmma_f32_16x16x32_bf16(false, a, false, b,
                                                 (short)0, c, false, false);
}

__device__ __forceinline__ v16bf cat8(v8bf lo, v8bf hi) {
  return __builtin_shufflevector(lo, hi, 0,1,2,3,4,5,6,7,8,9,10,11,12,13,14,15);
}

__device__ __forceinline__ v8bf zero8bf() {
  v8bf z;
#pragma unroll
  for (int i = 0; i < 8; ++i) z[i] = (__bf16)0.0f;
  return z;
}

__device__ __forceinline__ v8f zero8f() {
  v8f z;
#pragma unroll
  for (int i = 0; i < 8; ++i) z[i] = 0.0f;
  return z;
}

// ---- weight convert + transpose: Wt[n][k] = (bf16)W[k][n], 4 weights -------
__global__ __launch_bounds__(256)
void wt_cvt_transpose(const float* __restrict__ W0, const float* __restrict__ W1,
                      const float* __restrict__ W2, const float* __restrict__ W3,
                      bf16_t* __restrict__ out) {
  __shared__ float tile[32][33];
  const float* Ws[4] = {W0, W1, W2, W3};
  const float* W = Ws[blockIdx.z];
  bf16_t* O = out + (size_t)blockIdx.z * D_MODEL * D_MODEL;
  const int bx = blockIdx.x * 32;   // n tile
  const int by = blockIdx.y * 32;   // k tile
  const int tx = threadIdx.x, ty = threadIdx.y;   // 32 x 8
#pragma unroll
  for (int r = ty; r < 32; r += 8)
    tile[r][tx] = W[(size_t)(by + r) * D_MODEL + bx + tx];
  __syncthreads();
#pragma unroll
  for (int r = ty; r < 32; r += 8)
    O[(size_t)(bx + r) * D_MODEL + by + tx] = (bf16_t)tile[tx][r];
}

// ---- one projection: 32 rows, P = x @ W + b, bf16 result into LDS ----------
// MODE: 0 -> dstQ[p][c], 1 -> dstK[p][c], 2 -> V transposed dstV[p][d*16+j]
template<int MODE>
__device__ __forceinline__ void project32(
    const float* __restrict__ x, const bf16_t* __restrict__ Wt,
    const float* __restrict__ bias,
    bf16_t* S, bf16_t* dstQ, bf16_t* dstK, bf16_t* dstV,
    int row0, int wave, int lane)
{
  const int l16 = lane & 15, hlf = lane >> 4;
  const int tid = wave * 32 + lane;

  __syncthreads();                       // previous phase done with S
  // stage 32 rows of x (f32 -> bf16) into S[32][1024], coalesced
#pragma unroll
  for (int it = 0; it < (ROWS_PER_WG * D_MODEL) / THREADS; ++it) {
    const int i = it * THREADS + tid;
    S[i] = (bf16_t)x[(size_t)(row0 + (i >> 10)) * D_MODEL + (i & 1023)];
  }
  __syncthreads();

  v8f acc[2][4];
#pragma unroll
  for (int mt = 0; mt < 2; ++mt)
#pragma unroll
    for (int nt = 0; nt < 4; ++nt) acc[mt][nt] = zero8f();

  for (int k0 = 0; k0 < D_MODEL / 32; ++k0) {
    // A fragments (one per 16-row M tile): lane<16 holds K=0..7 & 16..23
    v16bf a[2];
#pragma unroll
    for (int mt = 0; mt < 2; ++mt) {
      const int ab = (mt * 16 + l16) * D_MODEL + k0 * 32 + hlf * 8;
      a[mt] = cat8(*(const v8bf*)&S[ab], *(const v8bf*)&S[ab + 16]);
    }
#pragma unroll
    for (int nt = 0; nt < 4; ++nt) {
      // B fragment: lane<16 holds K=0..7 & 16..23 of column (n=l16); lane>=16 K=8..15 & 24..31
      const int col = wave * 64 + nt * 16 + l16;
      const v8bf* bp = (const v8bf*)&Wt[(size_t)col * D_MODEL + k0 * 32 + hlf * 8];
      const v16bf b = cat8(bp[0], bp[2]);
      acc[0][nt] = wmma_bf16(a[0], b, acc[0][nt]);   // B reused by both M tiles
      acc[1][nt] = wmma_bf16(a[1], b, acc[1][nt]);
    }
  }

  // bias + writeback (C layout: VGPR r -> rows r / r+8, N = l16)
#pragma unroll
  for (int nt = 0; nt < 4; ++nt) {
    const int col = wave * 64 + nt * 16 + l16;
    const float bb = bias[col];
#pragma unroll
    for (int mt = 0; mt < 2; ++mt)
#pragma unroll
      for (int r = 0; r < 8; ++r) {
        const int rowp = mt * 16 + r + hlf * 8;
        const bf16_t h = (bf16_t)(acc[mt][nt][r] + bb);
        if (MODE == 0)      dstQ[rowp * D_MODEL + col] = h;
        else if (MODE == 1) dstK[rowp * D_MODEL + col] = h;
        else {              // V stored transposed per position: [p][d][j]
          const int j = col >> 6, d = col & 63;
          dstV[rowp * D_MODEL + d * 16 + j] = h;
        }
      }
  }
}

// ---- fused kernel ----------------------------------------------------------
__global__ __launch_bounds__(THREADS)
void mha_fused(const float* __restrict__ xq, const float* __restrict__ xk,
               const float* __restrict__ xv,
               const bf16_t* __restrict__ Wtq, const float* __restrict__ bq,
               const bf16_t* __restrict__ Wtk, const float* __restrict__ bk,
               const bf16_t* __restrict__ Wtv, const float* __restrict__ bv,
               const bf16_t* __restrict__ Wto, const float* __restrict__ bo,
               float* __restrict__ out)
{
  extern __shared__ char smem_raw[];
  bf16_t* S  = (bf16_t*)smem_raw;            // [32][1024] input staging (64 KB)
  bf16_t* Qm = S  + ROWS_PER_WG * D_MODEL;   // [32][1024] Q, later attn (64 KB)
  bf16_t* Km = Qm + ROWS_PER_WG * D_MODEL;   // [32][1024] K            (64 KB)
  bf16_t* Vt = Km + ROWS_PER_WG * D_MODEL;   // [32][64][16] V^T        (64 KB)
  bf16_t* Wb = Vt + ROWS_PER_WG * D_MODEL;   // [32][16][16] softmax w  (16 KB)

  const int tid  = threadIdx.x;
  const int lane = tid & 31;
  const int wave = tid >> 5;                 // 0..15
  const int l16  = lane & 15, hlf = lane >> 4;
  const int row0 = blockIdx.x * ROWS_PER_WG;

  project32<0>(xq, Wtq, bq, S, Qm, Km, Vt, row0, wave, lane);
  project32<1>(xk, Wtk, bk, S, Qm, Km, Vt, row0, wave, lane);
  project32<2>(xv, Wtv, bv, S, Qm, Km, Vt, row0, wave, lane);
  __syncthreads();   // Q/K/V columns written by all waves

  // ---- per-position attention over heads: 2 positions per wave ----
  const v8bf zb = zero8bf();
#pragma unroll
  for (int pi = 0; pi < 2; ++pi) {
    const int p = wave * 2 + pi;             // 0..31
    // score = Q_p(16x64) @ K_p^T / sqrt(64)  -> 16x16, K split 2x32
    v8f sc = zero8f();
#pragma unroll
    for (int kk = 0; kk < 2; ++kk) {
      const int qa = p * D_MODEL + l16 * HDIM + kk * 32;
      const v16bf a = cat8(*(const v8bf*)&Qm[qa + hlf * 8],
                           *(const v8bf*)&Qm[qa + hlf * 8 + 16]);
      const v16bf b = cat8(*(const v8bf*)&Km[qa + hlf * 8],
                           *(const v8bf*)&Km[qa + hlf * 8 + 16]);
      sc = wmma_bf16(a, b, sc);
    }
    // softmax over head axis j = columns = lanes within each 16-lane half
#pragma unroll
    for (int r = 0; r < 8; ++r) {
      const float xs = sc[r] * 0.125f;
      float m = xs;
#pragma unroll
      for (int o = 1; o < 16; o <<= 1) m = fmaxf(m, __shfl_xor(m, o));
      const float e = __expf(xs - m);
      float s = e;
#pragma unroll
      for (int o = 1; o < 16; o <<= 1) s += __shfl_xor(s, o);
      Wb[p * 256 + (r + hlf * 8) * 16 + l16] = (bf16_t)(e / s);
    }
    // re-layout w through LDS into an A fragment (K padded 16 -> 32 w/ zeros);
    // LDS ops from one wave complete in order, no barrier needed.
    const v16bf aw = cat8(*(const v8bf*)&Wb[p * 256 + l16 * 16 + hlf * 8], zb);
    v8f att[4];
#pragma unroll
    for (int nt = 0; nt < 4; ++nt) {
      const v8bf* vp = (const v8bf*)&Vt[p * D_MODEL + (nt * 16 + l16) * 16];
      v8bf b0 = vp[hlf], b1 = zb;         // K=16..31 are the zero padding
      att[nt] = wmma_bf16(aw, cat8(b0, b1), zero8f());
    }
    // attn rows (bf16) overwrite Qm -> A matrix of the output projection
#pragma unroll
    for (int nt = 0; nt < 4; ++nt)
#pragma unroll
      for (int r = 0; r < 8; ++r) {
        const int i = r + hlf * 8;         // head
        const int d = nt * 16 + l16;       // dim within head
        Qm[p * D_MODEL + i * HDIM + d] = (bf16_t)att[nt][r];
      }
  }
  __syncthreads();   // attn rows visible to all waves

  // ---- output projection: out = attn @ Wo + bo (f32 store) ----
  v8f acc[2][4];
#pragma unroll
  for (int mt = 0; mt < 2; ++mt)
#pragma unroll
    for (int nt = 0; nt < 4; ++nt) acc[mt][nt] = zero8f();

  for (int k0 = 0; k0 < D_MODEL / 32; ++k0) {
    v16bf a[2];
#pragma unroll
    for (int mt = 0; mt < 2; ++mt) {
      const int ab = (mt * 16 + l16) * D_MODEL + k0 * 32 + hlf * 8;
      a[mt] = cat8(*(const v8bf*)&Qm[ab], *(const v8bf*)&Qm[ab + 16]);
    }
#pragma unroll
    for (int nt = 0; nt < 4; ++nt) {
      const int col = wave * 64 + nt * 16 + l16;
      const v8bf* bp = (const v8bf*)&Wto[(size_t)col * D_MODEL + k0 * 32 + hlf * 8];
      const v16bf b = cat8(bp[0], bp[2]);
      acc[0][nt] = wmma_bf16(a[0], b, acc[0][nt]);
      acc[1][nt] = wmma_bf16(a[1], b, acc[1][nt]);
    }
  }
#pragma unroll
  for (int nt = 0; nt < 4; ++nt) {
    const int col = wave * 64 + nt * 16 + l16;
    const float bb = bo[col];
#pragma unroll
    for (int mt = 0; mt < 2; ++mt)
#pragma unroll
      for (int r = 0; r < 8; ++r)
        out[(size_t)(row0 + mt * 16 + r + hlf * 8) * D_MODEL + col] =
            acc[mt][nt][r] + bb;
  }
}

// ---------------------------------------------------------------------------
extern "C" void kernel_launch(void* const* d_in, const int* in_sizes, int n_in,
                              void* d_out, int out_size, void* d_ws, size_t ws_size,
                              hipStream_t stream) {
  const float* q  = (const float*)d_in[0];
  const float* k  = (const float*)d_in[1];
  const float* v  = (const float*)d_in[2];
  const float* Wq = (const float*)d_in[3];
  const float* bq = (const float*)d_in[4];
  const float* Wk = (const float*)d_in[5];
  const float* bk = (const float*)d_in[6];
  const float* Wv = (const float*)d_in[7];
  const float* bv = (const float*)d_in[8];
  const float* Wo = (const float*)d_in[9];
  const float* bo = (const float*)d_in[10];

  bf16_t* wt = (bf16_t*)d_ws;                      // 4 x 1024x1024 bf16 = 8 MB
  const bf16_t* Wtq = wt + (size_t)0 * D_MODEL * D_MODEL;
  const bf16_t* Wtk = wt + (size_t)1 * D_MODEL * D_MODEL;
  const bf16_t* Wtv = wt + (size_t)2 * D_MODEL * D_MODEL;
  const bf16_t* Wto = wt + (size_t)3 * D_MODEL * D_MODEL;

  dim3 tb(32, 8), tg(D_MODEL / 32, D_MODEL / 32, 4);
  wt_cvt_transpose<<<tg, tb, 0, stream>>>(Wq, Wk, Wv, Wo, wt);

  const int rows = 4 * 4096;                       // B * S
  const size_t smem =
      (size_t)(4 * ROWS_PER_WG * D_MODEL + ROWS_PER_WG * 256) * sizeof(bf16_t); // 272 KB
  mha_fused<<<rows / ROWS_PER_WG, THREADS, smem, stream>>>(
      q, k, v, Wtq, bq, Wtk, bk, Wtv, bv, Wto, bo, (float*)d_out);
}
